// TrajectoryModel_49838800503423
// MI455X (gfx1250) — compile-verified, matrix operand order checked by reading, not run on hardware
//
#include <hip/hip_runtime.h>
#include <hip/hip_bf16.h>
#include <stdint.h>

// ---------------------------------------------------------------------------
// MI455X (gfx1250, wave32) trajectory transformer.
// ~2.1e11 FLOP, matrix-op bound -> v_wmma_f32_16x16x32_f16 for all GEMMs +
// encoder attention.  Weights pre-transposed and zero-padded to f16 [Npad,Kpad]
// so every WMMA operand is a pair of unguarded 16B vector loads; V stored
// transposed per batch so attention A*V is vector-loaded too.  All matmul
// kernels are compile-time specialized (k_gemm_fast<KD>) -- no guarded
// scalar fragment path exists anywhere.
// ---------------------------------------------------------------------------

typedef __attribute__((ext_vector_type(16))) _Float16 v16h;
typedef __attribute__((ext_vector_type(8)))  _Float16 v8h;
typedef __attribute__((ext_vector_type(4)))  _Float16 v4h;
typedef __attribute__((ext_vector_type(8)))  float    v8f;

static constexpr int CB    = 1024;   // batch
static constexpr int CK    = 256;    // modes (encoder tokens)
static constexpr int CD    = 128;    // model dim
static constexpr int CH    = 4;      // heads
static constexpr int CHD   = 32;     // head dim
static constexpr int COBS  = 8;
static constexpr int CPRED = 12;
static constexpr int CINS  = 2;
static constexpr int CN    = 8;      // neighbors
static constexpr int CTK   = 3;      // NUM_TRAIN_K
static constexpr int CM    = CB * CK;        // 262144
static constexpr int CMF   = CB * CTK;       // 3072
static constexpr int CMN   = CB * CN;        // 8192
static constexpr int SEQF  = CINS * (COBS + CPRED); // 40
static constexpr int SEQP  = 64;                    // padded seq K
static constexpr int NEIF  = CINS * COBS;           // 16
static constexpr int NEIP  = 32;                    // padded nei K
static constexpr int REGN  = CINS * CPRED;          // 24
static constexpr int REGP  = 64;                    // padded reg N
static constexpr int PRED_ELEMS = CMF * REGN;       // 73728

// ---------------------------------------------------------------------------
// WMMA fragment helpers.  16-bit A-matrix 16x32 layout (ISA 7.12.2):
//   lane L: row = L&15 ; K-base kb = (L<16 ? 0 : 8)
//   frag elems 0..7 -> K = kb+0..7 ; elems 8..15 -> K = kb+16..23
// Both operands use this pattern (B from transposed weights Wt[N,K], row=n).
// ---------------------------------------------------------------------------

__device__ __forceinline__ v8f wmma16(v16h a, v16h b, v8f c) {
  return __builtin_amdgcn_wmma_f32_16x16x32_f16(false, a, false, b,
                                                (short)0, c, false, false);
}

// Unguarded fragment load: p = &row_base[kb]; two 16B loads.
__device__ __forceinline__ v16h ldfrag(const _Float16* p) {
  v8h lo = *(const v8h*)(p);        // K = kb .. kb+7
  v8h hi = *(const v8h*)(p + 16);   // K = kb+16 .. kb+23
  v16h r;
#pragma unroll
  for (int i = 0; i < 8; ++i) { r[i] = lo[i]; r[8 + i] = hi[i]; }
  return r;
}

// C/D tile: VGPR r -> M = row0 + r + (lane<16?0:8), N = n0 + (lane&15)
__device__ __forceinline__ void store_tile_fast(v8f acc, int row0, int n0,
                                                int N, const float* bias,
                                                int relu, float* Cf,
                                                _Float16* Ch, int vt) {
  const int lane = threadIdx.x;
  const int n    = n0 + (lane & 15);
  const int moff = (lane >> 4) << 3;
  const float bb = bias[n];
#pragma unroll
  for (int r = 0; r < 8; ++r) {
    const int m = row0 + moff + r;
    float v = acc[r] + bb;
    if (relu) v = fmaxf(v, 0.0f);
    if (Cf) Cf[(size_t)m * N + n] = v;
    if (Ch) {
      if (vt)  // per-batch transposed: VT[b][channel n][token m%256]
        Ch[((size_t)(m >> 8) * CD + n) * CK + (m & 255)] = (_Float16)v;
      else
        Ch[(size_t)m * N + n] = (_Float16)v;
    }
  }
}

// ---------------------------------------------------------------------------
// Fast WMMA GEMM (compile-time K): C[M,N] = A[M,KD] @ Wt[N,KD]^T (+bias)(+relu)
// Requires M%64==0, N%64==0.  block(32,4): wave ty owns row tile
// (blockIdx.x*4+ty)*16 and a 64-wide N strip; 1 A-frag feeds 4 WMMAs/k-step.
// Fully unrolled, immediate-offset vector loads, no guards near WMMA.
// ---------------------------------------------------------------------------
template <int KD>
__global__ void __launch_bounds__(128)
k_gemm_fast(const _Float16* __restrict__ A, const _Float16* __restrict__ Wt,
            const float* __restrict__ bias, float* __restrict__ Cf,
            _Float16* __restrict__ Ch, int N, int relu, int vt) {
  const int lane = threadIdx.x;
  const int row0 = (blockIdx.x * 4 + threadIdx.y) * 16;
  const int nb   = blockIdx.y * 64;
  const int kb   = (lane >> 4) << 3;
  const _Float16* ap = A  + (size_t)(row0 + (lane & 15)) * KD + kb;
  const _Float16* bp = Wt + (size_t)(nb   + (lane & 15)) * KD + kb;
  v8f acc0 = {}, acc1 = {}, acc2 = {}, acc3 = {};
#pragma unroll
  for (int k0 = 0; k0 < KD; k0 += 32) {
    v16h a  = ldfrag(ap + k0);
    v16h b0 = ldfrag(bp + k0);
    v16h b1 = ldfrag(bp + 16 * KD + k0);
    v16h b2 = ldfrag(bp + 32 * KD + k0);
    v16h b3 = ldfrag(bp + 48 * KD + k0);
    acc0 = wmma16(a, b0, acc0);
    acc1 = wmma16(a, b1, acc1);
    acc2 = wmma16(a, b2, acc2);
    acc3 = wmma16(a, b3, acc3);
  }
  store_tile_fast(acc0, row0, nb +  0, N, bias, relu, Cf, Ch, vt);
  store_tile_fast(acc1, row0, nb + 16, N, bias, relu, Cf, Ch, vt);
  store_tile_fast(acc2, row0, nb + 32, N, bias, relu, Cf, Ch, vt);
  store_tile_fast(acc3, row0, nb + 48, N, bias, relu, Cf, Ch, vt);
}

// ---------------------------------------------------------------------------
// Encoder self-attention: block per (16-query tile, head, batch).
// e = QK^T/sqrt(hd) (WMMA) -> LDS softmax -> O = A*V (WMMA, V pre-transposed)
// All fragment loads unguarded (provably in-bounds).
// ---------------------------------------------------------------------------
__global__ void __launch_bounds__(256)
k_attn_enc(const _Float16* __restrict__ Q, const _Float16* __restrict__ Kt,
           const _Float16* __restrict__ VT, _Float16* __restrict__ O) {
  __shared__ float    eS[16][260];
  __shared__ _Float16 aS[16][256];
  __shared__ float    red[16][17];
  const int lane = threadIdx.x, wave = threadIdx.y;
  const int qt = blockIdx.x, h = blockIdx.y, b = blockIdx.z;
  const int col   = h * CHD;
  const int kb    = (lane >> 4) << 3;
  const int qrow0 = b * CK + qt * 16;
  const float scale = 0.17677669529663687f; // 1/sqrt(32)

  v16h qf = ldfrag(Q + (size_t)(qrow0 + (lane & 15)) * CD + col + kb);

  for (int t = wave; t < 16; t += 8) {
    const int krow0 = b * CK + t * 16;
    v16h kf = ldfrag(Kt + (size_t)(krow0 + (lane & 15)) * CD + col + kb);
    v8f acc = {};
    acc = wmma16(qf, kf, acc);
    const int n    = t * 16 + (lane & 15);
    const int moff = (lane >> 4) << 3;
#pragma unroll
    for (int r = 0; r < 8; ++r) eS[moff + r][n] = acc[r] * scale;
  }
  __syncthreads();

  const int tid = wave * 32 + lane;
  const int rr = tid >> 4, cc = tid & 15;
  float mx = -3.0e38f;
  for (int j = cc; j < 256; j += 16) mx = fmaxf(mx, eS[rr][j]);
  red[rr][cc] = mx;
  __syncthreads();
  if (cc == 0) {
    float m2 = red[rr][0];
    for (int j = 1; j < 16; ++j) m2 = fmaxf(m2, red[rr][j]);
    red[rr][16] = m2;
  }
  __syncthreads();
  const float rowmax = red[rr][16];
  float sm = 0.0f;
  for (int j = cc; j < 256; j += 16) {
    float ev = __expf(eS[rr][j] - rowmax);
    eS[rr][j] = ev;
    sm += ev;
  }
  __syncthreads();
  red[rr][cc] = sm;
  __syncthreads();
  if (cc == 0) {
    float s2 = 0.0f;
    for (int j = 0; j < 16; ++j) s2 += red[rr][j];
    red[rr][16] = s2;
  }
  __syncthreads();
  const float inv = 1.0f / red[rr][16];
  for (int j = cc; j < 256; j += 16) aS[rr][j] = (_Float16)(eS[rr][j] * inv);
  __syncthreads();

  if (wave < 2) {
    const int n0 = wave * 16;
    const _Float16* VTb = VT + ((size_t)b * CD + col + n0 + (lane & 15)) * CK + kb;
    const _Float16* aP  = &aS[lane & 15][kb];
    v8f acc = {};
#pragma unroll
    for (int k0 = 0; k0 < 256; k0 += 32) {
      v16h af = ldfrag(aP + k0);    // LDS: ds_load_b128 pair
      v16h vf = ldfrag(VTb + k0);
      acc = wmma16(af, vf, acc);
    }
    const int n    = col + n0 + (lane & 15);
    const int moff = (lane >> 4) << 3;
#pragma unroll
    for (int r = 0; r < 8; ++r)
      O[(size_t)(qrow0 + moff + r) * CD + n] = (_Float16)acc[r];
  }
}

// ---------------------------------------------------------------------------
// Elementwise / small kernels
// ---------------------------------------------------------------------------
__global__ void k_pack_f16(const float* __restrict__ src,
                           _Float16* __restrict__ dst, long n4) {
  long i = (long)blockIdx.x * blockDim.x + threadIdx.x;
  long stride = (long)gridDim.x * blockDim.x;
  const float4* s4 = (const float4*)src;
  v4h* d4 = (v4h*)dst;
  for (; i < n4; i += stride) {
    float4 v = s4[i];
    v4h h;
    h[0] = (_Float16)v.x; h[1] = (_Float16)v.y;
    h[2] = (_Float16)v.z; h[3] = (_Float16)v.w;
    d4[i] = h;
  }
}

// dst[r*dstK + j] = j<srcK ? (f16)src[r*srcK + j] : 0   (row K-padding)
__global__ void k_pack_pad(const float* __restrict__ src,
                           _Float16* __restrict__ dst, long rows,
                           int srcK, int dstK) {
  long t = (long)blockIdx.x * blockDim.x + threadIdx.x;
  long stride = (long)gridDim.x * blockDim.x;
  const long total = rows * dstK;
  for (; t < total; t += stride) {
    int j = (int)(t % dstK);
    long r = t / dstK;
    dst[t] = (j < srcK) ? (_Float16)src[r * srcK + j] : (_Float16)0.0f;
  }
}

// Wt[n*Kpad+k] = (k<K && n<N) ? (f16)W[k*N+n] : 0
__global__ void k_transpose_w(const float* __restrict__ W,
                              _Float16* __restrict__ Wt, int K, int N,
                              int Kpad, int Npad) {
  long t = (long)blockIdx.x * blockDim.x + threadIdx.x;
  if (t >= (long)Npad * Kpad) return;
  int n = (int)(t / Kpad), k = (int)(t % Kpad);
  Wt[t] = (k < K && n < N) ? (_Float16)W[(size_t)k * N + n] : (_Float16)0.0f;
}

// out[j] = j<N ? b[j] : 0
__global__ void k_pad_bias(const float* __restrict__ b, float* __restrict__ out,
                           int N, int Npad) {
  int j = blockIdx.x * blockDim.x + threadIdx.x;
  if (j < Npad) out[j] = (j < N) ? b[j] : 0.0f;
}

// extract [CMF, REGN] from padded [CMF, REGP]
__global__ void k_copy_pred(const float* __restrict__ src,
                            float* __restrict__ dst) {
  long t = (long)blockIdx.x * blockDim.x + threadIdx.x;
  if (t >= (long)CMF * REGN) return;
  long m = t / REGN;
  int j = (int)(t % REGN);
  dst[t] = src[m * REGP + j];
}

// seq[b,k,:] = concat(ped_obs[b](16), motion_modes[k](24), zeros(24)) -> f16[64]
__global__ void k_build_seq(const float* __restrict__ ped,
                            const float* __restrict__ modes,
                            _Float16* __restrict__ S) {
  long i = (long)blockIdx.x * blockDim.x + threadIdx.x;
  long stride = (long)gridDim.x * blockDim.x;
  const long total = (long)CM * SEQP;
  for (; i < total; i += stride) {
    int j = (int)(i % SEQP);
    long m = i / SEQP;
    int b = (int)(m / CK), kk = (int)(m % CK);
    float v = 0.0f;
    if (j < CINS * COBS)      v = ped[(size_t)b * (CINS * COBS) + j];
    else if (j < SEQF)        v = modes[(size_t)kk * (CINS * CPRED) + (j - CINS * COBS)];
    S[i] = (_Float16)v;
  }
}

// X = LN(X + R) * g + b
__global__ void __launch_bounds__(128)
k_add_ln(float* __restrict__ X, const float* __restrict__ R,
         const float* __restrict__ g, const float* __restrict__ bb) {
  __shared__ float s1[128], s2[128];
  const int m = blockIdx.x, d = threadIdx.x;
  float x = X[(size_t)m * CD + d] + R[(size_t)m * CD + d];
  s1[d] = x; s2[d] = x * x;
  __syncthreads();
  for (int o = 64; o > 0; o >>= 1) {
    if (d < o) { s1[d] += s1[d + o]; s2[d] += s2[d + o]; }
    __syncthreads();
  }
  const float mean = s1[0] * (1.0f / CD);
  const float var  = s2[0] * (1.0f / CD) - mean * mean;
  X[(size_t)m * CD + d] = (x - mean) * rsqrtf(var + 1e-5f) * g[d] + bb[d];
}

__global__ void k_cls(const float* __restrict__ X, const float* __restrict__ W,
                      const float* __restrict__ b, float* __restrict__ out) {
  long m = (long)blockIdx.x * blockDim.x + threadIdx.x;
  if (m >= CM) return;
  float acc = b[0];
  for (int j = 0; j < CD; ++j) acc += X[(size_t)m * CD + j] * W[j];
  out[m] = acc;
}

__global__ void k_topk(const float* __restrict__ scores,
                       const int* __restrict__ closest, int* __restrict__ idx) {
  int b = blockIdx.x * blockDim.x + threadIdx.x;
  if (b >= CB) return;
  const float* s = scores + (size_t)b * CK;
  float v1 = -3.0e38f, v2 = -3.0e38f;
  int i1 = 0, i2 = 0;
  for (int j = 0; j < CK; ++j) {
    float v = s[j];
    if (v > v1)      { v2 = v1; i2 = i1; v1 = v; i1 = j; }
    else if (v > v2) { v2 = v;  i2 = j; }
  }
  idx[b * 3 + 0] = closest[b];
  idx[b * 3 + 1] = i1;
  idx[b * 3 + 2] = i2;
}

__global__ void k_gather(const float* __restrict__ X, const int* __restrict__ idx,
                         float* __restrict__ feat) {
  long t = (long)blockIdx.x * blockDim.x + threadIdx.x;
  const long total = (long)CMF * CD;
  if (t >= total) return;
  int d = (int)(t % CD);
  long row = t / CD;
  int b = (int)(row / CTK), j = (int)(row % CTK);
  feat[t] = X[((size_t)b * CK + idx[b * 3 + j]) * CD + d];
}

// Decoder cross-attention: 3 queries x 8 keys, scalar (tiny).
__global__ void k_attn_dec(const float* __restrict__ q, const float* __restrict__ k,
                           const float* __restrict__ v, const float* __restrict__ mask,
                           float* __restrict__ o) {
  int t = blockIdx.x * blockDim.x + threadIdx.x;
  if (t >= CB * CTK * CH) return;
  const int h  = t % CH;
  const int tq = (t / CH) % CTK;
  const int b  = t / (CH * CTK);
  const float* qp = q + ((size_t)b * CTK + tq) * CD + h * CHD;
  float sc[CN]; float mx = -3.0e38f;
  for (int n = 0; n < CN; ++n) {
    const float* kp = k + ((size_t)b * CN + n) * CD + h * CHD;
    float d = 0.0f;
    for (int c = 0; c < CHD; ++c) d += qp[c] * kp[c];
    d *= 0.17677669529663687f;
    if (mask[(size_t)b * CN * CN + n] <= 0.0f) d = -1e9f; // mask[b,0,n]
    sc[n] = d; mx = fmaxf(mx, d);
  }
  float s = 0.0f;
  for (int n = 0; n < CN; ++n) { sc[n] = __expf(sc[n] - mx); s += sc[n]; }
  const float inv = 1.0f / s;
  float out[CHD];
  for (int c = 0; c < CHD; ++c) out[c] = 0.0f;
  for (int n = 0; n < CN; ++n) {
    const float w = sc[n] * inv;
    const float* vp = v + ((size_t)b * CN + n) * CD + h * CHD;
    for (int c = 0; c < CHD; ++c) out[c] += w * vp[c];
  }
  float* op = o + ((size_t)b * CTK + tq) * CD + h * CHD;
  for (int c = 0; c < CHD; ++c) op[c] = out[c];
}

// ---------------------------------------------------------------------------
// Host side
// ---------------------------------------------------------------------------
struct LayerP {
  const float *qW, *qb, *kW, *kb, *vW, *vb, *oW, *ob;
  const float *f1W, *f1b, *f2W, *f2b;
  const float *ln1g, *ln1b, *ln2g, *ln2b;
};
struct LayerT {
  const _Float16 *q, *k, *v, *o, *f1, *f2;
  const float *qb, *kb, *vb, *ob, *f1b, *f2b;
  const float *ln1g, *ln1b, *ln2g, *ln2b;
};

// jax sorted-key layer leaves: ff1.W ff1.b ff2.W ff2.b k.W k.b ln1_b ln1_g
//                              ln2_b ln2_g o.W o.b q.W q.b v.W v.b
static LayerP get_layer(void* const* d, int base) {
  LayerP l;
  l.f1W = (const float*)d[base + 0];  l.f1b = (const float*)d[base + 1];
  l.f2W = (const float*)d[base + 2];  l.f2b = (const float*)d[base + 3];
  l.kW  = (const float*)d[base + 4];  l.kb  = (const float*)d[base + 5];
  l.ln1b = (const float*)d[base + 6]; l.ln1g = (const float*)d[base + 7];
  l.ln2b = (const float*)d[base + 8]; l.ln2g = (const float*)d[base + 9];
  l.oW  = (const float*)d[base + 10]; l.ob  = (const float*)d[base + 11];
  l.qW  = (const float*)d[base + 12]; l.qb  = (const float*)d[base + 13];
  l.vW  = (const float*)d[base + 14]; l.vb  = (const float*)d[base + 15];
  return l;
}

extern "C" void kernel_launch(void* const* d_in, const int* in_sizes, int n_in,
                              void* d_out, int out_size, void* d_ws, size_t ws_size,
                              hipStream_t stream) {
  (void)in_sizes; (void)n_in; (void)out_size; (void)ws_size;
  // ---- inputs (jax sorted-key tree flatten of setup_inputs dict; 77 leaves)
  const int*   closest = (const int*)d_in[0];
  const float* mask    = (const float*)d_in[1];
  const float* modes   = (const float*)d_in[2];
  const float* neis    = (const float*)d_in[3];
  const float* ped     = (const float*)d_in[4];
  const float* clsW    = (const float*)d_in[5];
  const float* clsB    = (const float*)d_in[6];
  LayerP dec0p = get_layer(d_in, 7);
  LayerP dec1p = get_layer(d_in, 23);
  const float* embW = (const float*)d_in[39];
  const float* embB = (const float*)d_in[40];
  LayerP enc0p = get_layer(d_in, 41);
  LayerP enc1p = get_layer(d_in, 57);
  const float* neiW = (const float*)d_in[73];
  const float* neiB = (const float*)d_in[74];
  const float* regW = (const float*)d_in[75];
  const float* regB = (const float*)d_in[76];

  // ---- workspace layout (~662 MB)
  char* ws = (char*)d_ws;
  float*    X     = (float*)   (ws + 0);            // f32 [M,128]
  _Float16* XH    = (_Float16*)(ws + 134217728);    // f16 [M,128] (also seq16[M,64])
  _Float16* A16   = (_Float16*)(ws + 201326592);    // f16 [M,256] (attn-out/hidden)
  _Float16* QH    = (_Float16*)(ws + 335544320);    // f16 [M,128]
  _Float16* KH    = (_Float16*)(ws + 402653184);    // f16 [M,128]
  _Float16* VT    = (_Float16*)(ws + 469762048);    // f16 [B,128,256] V transposed
  float*    T2    = (float*)   (ws + 536870912);    // f32 [M,128]
  float*    NEI   = (float*)   (ws + 671088640);    // f32 [8192,128]
  _Float16* NEI16 = (_Float16*)(ws + 675282944);    // f16 [8192,128]
  float*    FEAT  = (float*)   (ws + 677380096);    // f32 [3072,128]
  float*    FQ    = (float*)   (ws + 678952960);
  float*    FK    = (float*)   (ws + 680525824);    // f32 [8192,128]
  float*    FV    = (float*)   (ws + 684720128);
  float*    FO    = (float*)   (ws + 688914432);
  float*    DT2   = (float*)   (ws + 690487296);
  int*      IDX   = (int*)     (ws + 692060160);    // int [B,3]
  float*    PREDP = (float*)   (ws + 692072448);    // f32 [3072,64] padded pred
  float*    REGBP = (float*)   (ws + 692858880);    // f32 [64] padded reg bias
  _Float16* WT    = (_Float16*)(ws + 692859136);    // f16 transposed weight arena

  float* predOut   = (float*)d_out;
  float* scoresOut = (float*)d_out + PRED_ELEMS;

  // ---- pre-transpose all weights to f16 [Npad,Kpad] (zero-padded)
  long wtoff = 0;
  auto xp = [&](const float* W, int K, int N, int Kpad, int Npad) -> const _Float16* {
    _Float16* p = WT + wtoff;
    wtoff += (long)Npad * Kpad;
    long tot = (long)Npad * Kpad;
    k_transpose_w<<<(unsigned)((tot + 255) / 256), 256, 0, stream>>>(W, p, K, N,
                                                                     Kpad, Npad);
    return p;
  };
  auto xlayer = [&](const LayerP& s) -> LayerT {
    LayerT t;
    t.q = xp(s.qW, CD, CD, CD, CD);               t.qb = s.qb;
    t.k = xp(s.kW, CD, CD, CD, CD);               t.kb = s.kb;
    t.v = xp(s.vW, CD, CD, CD, CD);               t.vb = s.vb;
    t.o = xp(s.oW, CD, CD, CD, CD);               t.ob = s.ob;
    t.f1 = xp(s.f1W, CD, 2 * CD, CD, 2 * CD);     t.f1b = s.f1b;
    t.f2 = xp(s.f2W, 2 * CD, CD, 2 * CD, CD);     t.f2b = s.f2b;
    t.ln1g = s.ln1g; t.ln1b = s.ln1b; t.ln2g = s.ln2g; t.ln2b = s.ln2b;
    return t;
  };
  LayerT enc0 = xlayer(enc0p), enc1 = xlayer(enc1p);
  LayerT dec0 = xlayer(dec0p), dec1 = xlayer(dec1p);
  const _Float16* embT = xp(embW, SEQF, CD, SEQP, CD);    // [128,64]
  const _Float16* neiT = xp(neiW, NEIF, CD, NEIP, CD);    // [128,32]
  const _Float16* regT = xp(regW, CD, REGN, CD, REGP);    // [64,128] zero rows
  k_pad_bias<<<1, REGP, 0, stream>>>(regB, REGBP, REGN, REGP);

  // fast GEMM dispatch (M%64==0, N%64==0, K in {32,64,128,256})
  auto gemmF = [&](const _Float16* A, const _Float16* Wt, const float* bias,
                   float* Cf, _Float16* Ch, int M_, int K_, int N_, int relu,
                   int vt) {
    dim3 g(M_ / 64, N_ / 64);
    dim3 blk(32, 4);
    switch (K_) {
      case 32:  k_gemm_fast<32><<<g, blk, 0, stream>>>(A, Wt, bias, Cf, Ch, N_, relu, vt); break;
      case 64:  k_gemm_fast<64><<<g, blk, 0, stream>>>(A, Wt, bias, Cf, Ch, N_, relu, vt); break;
      case 128: k_gemm_fast<128><<<g, blk, 0, stream>>>(A, Wt, bias, Cf, Ch, N_, relu, vt); break;
      default:  k_gemm_fast<256><<<g, blk, 0, stream>>>(A, Wt, bias, Cf, Ch, N_, relu, vt); break;
    }
  };
  auto pack = [&](const float* s, _Float16* d, long n) {
    k_pack_f16<<<4096, 256, 0, stream>>>(s, d, n / 4);
  };
  auto addln = [&](float* Xp, const float* Rp, const float* g, const float* b,
                   int M_) {
    k_add_ln<<<M_, 128, 0, stream>>>(Xp, Rp, g, b);
  };

  auto enc_block = [&](const LayerT& L) {
    pack(X, XH, (long)CM * CD);
    gemmF(XH, L.q, L.qb, nullptr, QH, CM, CD, CD, 0, 0);
    gemmF(XH, L.k, L.kb, nullptr, KH, CM, CD, CD, 0, 0);
    gemmF(XH, L.v, L.vb, nullptr, VT, CM, CD, CD, 0, 1);   // store transposed
    k_attn_enc<<<dim3(16, CH, CB), dim3(32, 8), 0, stream>>>(QH, KH, VT, A16);
    gemmF(A16, L.o, L.ob, T2, nullptr, CM, CD, CD, 0, 0);
    addln(X, T2, L.ln1g, L.ln1b, CM);
    pack(X, XH, (long)CM * CD);
    gemmF(XH, L.f1, L.f1b, nullptr, A16, CM, CD, 2 * CD, 1, 0);  // relu, f16
    gemmF(A16, L.f2, L.f2b, T2, nullptr, CM, 2 * CD, CD, 0, 0);
    addln(X, T2, L.ln2g, L.ln2b, CM);
  };

  auto dec_block = [&](const LayerT& L) {
    pack(FEAT, XH, (long)CMF * CD);
    gemmF(XH, L.q, L.qb, FQ, nullptr, CMF, CD, CD, 0, 0);
    gemmF(NEI16, L.k, L.kb, FK, nullptr, CMN, CD, CD, 0, 0);
    gemmF(NEI16, L.v, L.vb, FV, nullptr, CMN, CD, CD, 0, 0);
    k_attn_dec<<<(CB * CTK * CH + 255) / 256, 256, 0, stream>>>(FQ, FK, FV,
                                                                mask, FO);
    pack(FO, XH, (long)CMF * CD);
    gemmF(XH, L.o, L.ob, DT2, nullptr, CMF, CD, CD, 0, 0);
    addln(FEAT, DT2, L.ln1g, L.ln1b, CMF);
    pack(FEAT, XH, (long)CMF * CD);
    gemmF(XH, L.f1, L.f1b, nullptr, A16, CMF, CD, 2 * CD, 1, 0);
    gemmF(A16, L.f2, L.f2b, DT2, nullptr, CMF, 2 * CD, CD, 0, 0);
    addln(FEAT, DT2, L.ln2g, L.ln2b, CMF);
  };

  // ---- 1. embed: seq16[M,64] -> X
  k_build_seq<<<4096, 256, 0, stream>>>(ped, modes, XH);
  gemmF(XH, embT, embB, X, nullptr, CM, SEQP, CD, 0, 0);

  // ---- 2. encoder (2 layers)
  enc_block(enc0);
  enc_block(enc1);

  // ---- 3. scores -> d_out tail; top-k; gather feat
  k_cls<<<(CM + 255) / 256, 256, 0, stream>>>(X, clsW, clsB, scoresOut);
  k_topk<<<(CB + 255) / 256, 256, 0, stream>>>(scoresOut, closest, IDX);
  k_gather<<<(unsigned)(((long)CMF * CD + 255) / 256), 256, 0, stream>>>(X, IDX,
                                                                         FEAT);

  // ---- 4. neighbor embed: padded [8192,32]f16 @ Wt[128,32] -> NEI
  k_pack_pad<<<2048, 256, 0, stream>>>(neis, NEI16, CMN, NEIF, NEIP);
  gemmF(NEI16, neiT, neiB, NEI, nullptr, CMN, NEIP, CD, 0, 0);
  pack(NEI, NEI16, (long)CMN * CD);

  // ---- 5. decoder: NUM_REFINE x (dec0, dec1)
  dec_block(dec0);
  dec_block(dec1);
  dec_block(dec0);
  dec_block(dec1);

  // ---- 6. regression head: padded N=64 fast GEMM, then extract 24 cols
  pack(FEAT, XH, (long)CMF * CD);
  gemmF(XH, regT, REGBP, PREDP, nullptr, CMF, CD, REGP, 0, 0);
  k_copy_pred<<<(unsigned)(((long)CMF * REGN + 255) / 256), 256, 0, stream>>>(
      PREDP, predOut);
}
